// LongT5LayerGraphSelfAttention_39298950758938
// MI455X (gfx1250) — compile-verified, matrix operand order checked by reading, not run on hardware
//
#include <hip/hip_runtime.h>

// ---------------------------------------------------------------------------
// T5 self-attention layer for MI455X (gfx1250, wave32, WMMA f32<-f16 16x16x32)
// ---------------------------------------------------------------------------

typedef __attribute__((ext_vector_type(16))) _Float16 v16h;
typedef __attribute__((ext_vector_type(8)))  _Float16 v8h;
typedef __attribute__((ext_vector_type(8)))  float    v8f;
typedef __attribute__((ext_vector_type(4)))  float    v4f;

#define D_MODEL 1024
#define SEQ     2048
#define BATCH   4
#define NHEADS  16
#define DKV     64

// ---- WMMA fragment helpers (CDNA5 ISA 7.12.2 layouts, wave32) --------------
// A (16x32 f16): lane l<16 -> row M=l, K halves {k0..k0+7, k0+16..k0+23};
//                lane l>=16 -> row M=l-16, K halves {k0+8.., k0+24..}.
// B (32x16 f16): mirrored with N per lane. C/D: VGPR r, row = r + 8*hi, col = l.
static __device__ __forceinline__ v16h load_frag(const _Float16* base, int ld,
                                                 int row0, int k0, int l, int hi) {
  const _Float16* p = base + (size_t)(row0 + l) * ld + (k0 + hi * 8);
  union { v16h v; v8h q[2]; } u;
  u.q[0] = *(const v8h*)(p);
  u.q[1] = *(const v8h*)(p + 16);
  return u.v;
}

static __device__ __forceinline__ v8f wmma_f16(v16h a, v16h b, v8f c) {
  return __builtin_amdgcn_wmma_f32_16x16x32_f16(false, a, false, b, (short)0, c,
                                                false, false);
}

static __device__ __forceinline__ v8f zero8() {
  v8f z;
#pragma unroll
  for (int r = 0; r < 8; ++r) z[r] = 0.0f;
  return z;
}

// ---- 1. T5 RMS layer norm -> f16 -------------------------------------------
__global__ __launch_bounds__(256) void ln_kernel(const float* __restrict__ x,
                                                 const float* __restrict__ w,
                                                 _Float16* __restrict__ xln) {
  __shared__ float red[256];
  int row = blockIdx.x;
  int t = threadIdx.x;
  const float* xr = x + (size_t)row * D_MODEL;
  float v[4];
  float s = 0.f;
#pragma unroll
  for (int i = 0; i < 4; ++i) { v[i] = xr[t + 256 * i]; s += v[i] * v[i]; }
  red[t] = s;
  __syncthreads();
  for (int o = 128; o > 0; o >>= 1) {
    if (t < o) red[t] += red[t + o];
    __syncthreads();
  }
  float inv = rsqrtf(red[0] * (1.0f / D_MODEL) + 1e-6f);
  _Float16* yr = xln + (size_t)row * D_MODEL;
#pragma unroll
  for (int i = 0; i < 4; ++i)
    yr[t + 256 * i] = (_Float16)(v[i] * inv * w[t + 256 * i]);
}

// ---- 2. weight transpose + f32 -> f16 (W[k][n] -> WT[n][k]) ----------------
__global__ __launch_bounds__(256) void wt_kernel(const float* __restrict__ src,
                                                 _Float16* __restrict__ dst) {
  int idx = (blockIdx.x * 256 + threadIdx.x) * 4;  // over 1024*1024
  int r = idx >> 10, c = idx & 1023;
  v4f vv = *(const v4f*)&src[idx];
#pragma unroll
  for (int i = 0; i < 4; ++i) dst[(size_t)(c + i) * 1024 + r] = (_Float16)vv[i];
}

// ---- 3a. per-head bias table over delta = k - q ----------------------------
__global__ __launch_bounds__(256) void bias_table_kernel(const float* __restrict__ rel,
                                                         float* __restrict__ table) {
  int idx = blockIdx.x * 256 + threadIdx.x;  // NHEADS * 4096
  int h = idx >> 12;
  int delta = (idx & 4095) - 2048;
  int bucket = (delta > 0) ? 16 : 0;  // num_buckets/2 = 16
  int rp = delta < 0 ? -delta : delta;
  if (rp < 8) {
    bucket += rp;
  } else {
    int lb = 8 + (int)(logf((float)rp * 0.125f) * (8.0f / logf(16.0f)));
    bucket += (lb < 15 ? lb : 15);
  }
  table[idx] = rel[bucket * NHEADS + h];
}

// ---- 3b. expand table to full [1,H,S,S] output (pure bandwidth) ------------
__global__ __launch_bounds__(256) void bias_out_kernel(const float* __restrict__ table,
                                                       float* __restrict__ bout) {
  size_t gid = ((size_t)blockIdx.x * 256 + threadIdx.x) * 4;
  int h = (int)(gid >> 22);
  int rem = (int)(gid & 4194303);
  int q = rem >> 11, k = rem & 2047;
  const float* th = table + h * 4096;
  v4f vv;
#pragma unroll
  for (int i = 0; i < 4; ++i) vv[i] = th[k + i - q + 2048];
  *(v4f*)&bout[gid] = vv;
}

// ---- 4. LDS-staged WMMA GEMM: C_f16[M,N] = A_f16[M,K] @ BT_f16[N,K]^T ------
__global__ __launch_bounds__(256) void gemm_f16_kernel(const _Float16* __restrict__ A,
                                                       const _Float16* __restrict__ BT,
                                                       _Float16* __restrict__ C,
                                                       int M, int N, int K) {
  __shared__ _Float16 sA[128][40];
  __shared__ _Float16 sB[128][40];
  int t = threadIdx.x, lane = t & 31, wave = t >> 5;
  int l = lane & 15, hi = lane >> 4;
  int m0 = blockIdx.x * 128, n0 = blockIdx.y * 128;
  int wm = wave >> 1, wn = wave & 1;  // 4x2 wave grid, 32x64 per wave
  v8f acc[2][4];
#pragma unroll
  for (int mi = 0; mi < 2; ++mi)
#pragma unroll
    for (int ni = 0; ni < 4; ++ni) acc[mi][ni] = zero8();

  for (int k0 = 0; k0 < K; k0 += 32) {
    __syncthreads();
#pragma unroll
    for (int c = 0; c < 2; ++c) {
      int idx = t + 256 * c;
      int row = idx >> 2, cc = idx & 3;
      *(v8h*)&sA[row][cc * 8] = *(const v8h*)&A[(size_t)(m0 + row) * K + k0 + cc * 8];
      *(v8h*)&sB[row][cc * 8] = *(const v8h*)&BT[(size_t)(n0 + row) * K + k0 + cc * 8];
      if (k0 + 32 < K) {
        __builtin_prefetch(&A[(size_t)(m0 + row) * K + k0 + 32 + cc * 8], 0, 1);
        __builtin_prefetch(&BT[(size_t)(n0 + row) * K + k0 + 32 + cc * 8], 0, 1);
      }
    }
    __syncthreads();
    v16h af[2];
#pragma unroll
    for (int mi = 0; mi < 2; ++mi)
      af[mi] = load_frag(&sA[0][0], 40, wm * 32 + mi * 16, 0, l, hi);
#pragma unroll
    for (int ni = 0; ni < 4; ++ni) {
      v16h bf = load_frag(&sB[0][0], 40, wn * 64 + ni * 16, 0, l, hi);
#pragma unroll
      for (int mi = 0; mi < 2; ++mi) acc[mi][ni] = wmma_f16(af[mi], bf, acc[mi][ni]);
    }
  }
#pragma unroll
  for (int mi = 0; mi < 2; ++mi)
#pragma unroll
    for (int ni = 0; ni < 4; ++ni)
#pragma unroll
      for (int r = 0; r < 8; ++r) {
        int row = m0 + wm * 32 + mi * 16 + r + 8 * hi;
        int col = n0 + wn * 64 + ni * 16 + l;
        C[(size_t)row * N + col] = (_Float16)acc[mi][ni][r];
      }
}

// ---- 6. final GEMM with fused residual, f32 output -------------------------
__global__ __launch_bounds__(256) void gemm_out_kernel(const _Float16* __restrict__ A,
                                                       const _Float16* __restrict__ BT,
                                                       const float* __restrict__ resid,
                                                       float* __restrict__ C,
                                                       int M, int N, int K) {
  __shared__ _Float16 sA[128][40];
  __shared__ _Float16 sB[128][40];
  int t = threadIdx.x, lane = t & 31, wave = t >> 5;
  int l = lane & 15, hi = lane >> 4;
  int m0 = blockIdx.x * 128, n0 = blockIdx.y * 128;
  int wm = wave >> 1, wn = wave & 1;
  v8f acc[2][4];
#pragma unroll
  for (int mi = 0; mi < 2; ++mi)
#pragma unroll
    for (int ni = 0; ni < 4; ++ni) acc[mi][ni] = zero8();

  for (int k0 = 0; k0 < K; k0 += 32) {
    __syncthreads();
#pragma unroll
    for (int c = 0; c < 2; ++c) {
      int idx = t + 256 * c;
      int row = idx >> 2, cc = idx & 3;
      *(v8h*)&sA[row][cc * 8] = *(const v8h*)&A[(size_t)(m0 + row) * K + k0 + cc * 8];
      *(v8h*)&sB[row][cc * 8] = *(const v8h*)&BT[(size_t)(n0 + row) * K + k0 + cc * 8];
    }
    __syncthreads();
    v16h af[2];
#pragma unroll
    for (int mi = 0; mi < 2; ++mi)
      af[mi] = load_frag(&sA[0][0], 40, wm * 32 + mi * 16, 0, l, hi);
#pragma unroll
    for (int ni = 0; ni < 4; ++ni) {
      v16h bf = load_frag(&sB[0][0], 40, wn * 64 + ni * 16, 0, l, hi);
#pragma unroll
      for (int mi = 0; mi < 2; ++mi) acc[mi][ni] = wmma_f16(af[mi], bf, acc[mi][ni]);
    }
  }
#pragma unroll
  for (int mi = 0; mi < 2; ++mi)
#pragma unroll
    for (int ni = 0; ni < 4; ++ni)
#pragma unroll
      for (int r = 0; r < 8; ++r) {
        int row = m0 + wm * 32 + mi * 16 + r + 8 * hi;
        int col = n0 + wn * 64 + ni * 16 + l;
        C[(size_t)row * N + col] = acc[mi][ni][r] + resid[(size_t)row * N + col];
      }
}

// ---- 5. flash attention: scores + bias + online softmax + P@V --------------
// Block: 4 waves / 128 threads, 128 q-rows per block (32 per wave).
// Grid: (SEQ/128, NHEADS, BATCH).
__global__ __launch_bounds__(128) void attn_kernel(const _Float16* __restrict__ Qb,
                                                   const _Float16* __restrict__ Kb,
                                                   const _Float16* __restrict__ Vb,
                                                   const float* __restrict__ table,
                                                   _Float16* __restrict__ Ctx) {
  __shared__ float tbl[4096];          // per-head bias over delta
  __shared__ _Float16 vt[64][72];      // V tile transposed: [d][key]
  __shared__ _Float16 pst[4][32][72];  // per-wave P staging (C->A relayout)
  int t = threadIdx.x, lane = t & 31, wave = t >> 5;
  int l = lane & 15, hi = lane >> 4;
  int qt = blockIdx.x, h = blockIdx.y, b = blockIdx.z;
  int q0 = qt * 128 + wave * 32;
  const _Float16* Qp = Qb + (size_t)b * SEQ * D_MODEL + h * DKV;
  const _Float16* Kp = Kb + (size_t)b * SEQ * D_MODEL + h * DKV;
  const _Float16* Vp = Vb + (size_t)b * SEQ * D_MODEL + h * DKV;

#pragma unroll
  for (int i = 0; i < 32; ++i) tbl[t + 128 * i] = table[h * 4096 + t + 128 * i];

  // Q fragments resident in registers for all 32 KV iterations
  v16h qf[2][2];
#pragma unroll
  for (int mi = 0; mi < 2; ++mi)
#pragma unroll
    for (int kc = 0; kc < 2; ++kc)
      qf[mi][kc] = load_frag(Qp, D_MODEL, q0 + mi * 16, kc * 32, l, hi);

  v8f o[2][4];
  float mst[2][8], lst[2][8];
#pragma unroll
  for (int mi = 0; mi < 2; ++mi) {
#pragma unroll
    for (int di = 0; di < 4; ++di) o[mi][di] = zero8();
#pragma unroll
    for (int r = 0; r < 8; ++r) { mst[mi][r] = -1e30f; lst[mi][r] = 0.0f; }
  }

  for (int j = 0; j < SEQ / 64; ++j) {
    int key0 = j * 64;
    __syncthreads();  // prior iteration done with vt
    // cooperative transposed V-tile stage: vt[d][key]
#pragma unroll
    for (int c = 0; c < 4; ++c) {
      int idx = t + 128 * c;  // 512 chunks of 8 halves
      int key = idx >> 3, dc = idx & 7;
      v8h vv = *(const v8h*)&Vp[(size_t)(key0 + key) * D_MODEL + dc * 8];
#pragma unroll
      for (int e = 0; e < 8; ++e) vt[dc * 8 + e][key] = vv[e];
    }
    __syncthreads();

    // scores S = Q @ K^T  (B fragment reads K rows: col=key, K-dim=d)
    v8f sc[2][4];
#pragma unroll
    for (int ni = 0; ni < 4; ++ni) {
      v16h kf0 = load_frag(Kp, D_MODEL, key0 + ni * 16, 0, l, hi);
      v16h kf1 = load_frag(Kp, D_MODEL, key0 + ni * 16, 32, l, hi);
#pragma unroll
      for (int mi = 0; mi < 2; ++mi) {
        v8f s = zero8();
        s = wmma_f16(qf[mi][0], kf0, s);
        s = wmma_f16(qf[mi][1], kf1, s);
        sc[mi][ni] = s;
      }
    }

    // bias add + online softmax update
#pragma unroll
    for (int mi = 0; mi < 2; ++mi) {
#pragma unroll
      for (int ni = 0; ni < 4; ++ni)
#pragma unroll
        for (int r = 0; r < 8; ++r) {
          int row = q0 + mi * 16 + r + 8 * hi;
          int col = key0 + ni * 16 + l;
          sc[mi][ni][r] += tbl[col - row + 2048];
        }
      float pm[8], sca[8], rs[8];
#pragma unroll
      for (int r = 0; r < 8; ++r) {
        float m4 = fmaxf(fmaxf(sc[mi][0][r], sc[mi][1][r]),
                         fmaxf(sc[mi][2][r], sc[mi][3][r]));
        m4 = fmaxf(m4, __shfl_xor(m4, 1, 32));
        m4 = fmaxf(m4, __shfl_xor(m4, 2, 32));
        m4 = fmaxf(m4, __shfl_xor(m4, 4, 32));
        m4 = fmaxf(m4, __shfl_xor(m4, 8, 32));
        pm[r] = m4;
      }
#pragma unroll
      for (int r = 0; r < 8; ++r) {
        float mnew = fmaxf(mst[mi][r], pm[r]);
        sca[r] = __expf(mst[mi][r] - mnew);
        mst[mi][r] = mnew;
        float s = 0.f;
#pragma unroll
        for (int ni = 0; ni < 4; ++ni) {
          float pv = __expf(sc[mi][ni][r] - mnew);
          sc[mi][ni][r] = pv;
          s += pv;
        }
        s += __shfl_xor(s, 1, 32);
        s += __shfl_xor(s, 2, 32);
        s += __shfl_xor(s, 4, 32);
        s += __shfl_xor(s, 8, 32);
        rs[r] = s;
      }
#pragma unroll
      for (int r = 0; r < 8; ++r) {
        lst[mi][r] = lst[mi][r] * sca[r] + rs[r];
#pragma unroll
        for (int di = 0; di < 4; ++di) o[mi][di][r] *= sca[r];
      }
      // stage P through LDS to re-layout C -> A fragments
#pragma unroll
      for (int ni = 0; ni < 4; ++ni)
#pragma unroll
        for (int r = 0; r < 8; ++r)
          pst[wave][mi * 16 + r + 8 * hi][ni * 16 + l] = (_Float16)sc[mi][ni][r];
    }

    // O += P @ V  (B fragment from transposed V tile: col=d, K-dim=key)
    v16h pf[2][2];
#pragma unroll
    for (int mi = 0; mi < 2; ++mi)
#pragma unroll
      for (int kc = 0; kc < 2; ++kc)
        pf[mi][kc] = load_frag(&pst[wave][0][0], 72, mi * 16, kc * 32, l, hi);
#pragma unroll
    for (int di = 0; di < 4; ++di)
#pragma unroll
      for (int kc = 0; kc < 2; ++kc) {
        v16h vf = load_frag(&vt[0][0], 72, di * 16, kc * 32, l, hi);
#pragma unroll
        for (int mi = 0; mi < 2; ++mi)
          o[mi][di] = wmma_f16(pf[mi][kc], vf, o[mi][di]);
      }
  }

  // normalize and store context (input to output projection)
#pragma unroll
  for (int mi = 0; mi < 2; ++mi) {
    float invl[8];
#pragma unroll
    for (int r = 0; r < 8; ++r) invl[r] = 1.0f / lst[mi][r];
#pragma unroll
    for (int di = 0; di < 4; ++di)
#pragma unroll
      for (int r = 0; r < 8; ++r) {
        int row = q0 + mi * 16 + r + 8 * hi;
        int col = h * DKV + di * 16 + l;
        Ctx[((size_t)b * SEQ + row) * D_MODEL + col] =
            (_Float16)(o[mi][di][r] * invl[r]);
      }
  }
}

// ---------------------------------------------------------------------------
extern "C" void kernel_launch(void* const* d_in, const int* in_sizes, int n_in,
                              void* d_out, int out_size, void* d_ws, size_t ws_size,
                              hipStream_t stream) {
  (void)in_sizes; (void)n_in; (void)out_size; (void)ws_size;
  const float* hs  = (const float*)d_in[0];
  const float* lnw = (const float*)d_in[1];
  const float* Wq  = (const float*)d_in[2];
  const float* Wk  = (const float*)d_in[3];
  const float* Wv  = (const float*)d_in[4];
  const float* Wo  = (const float*)d_in[5];
  const float* rel = (const float*)d_in[6];

  float* out = (float*)d_out;
  float* bias_out = out + (size_t)BATCH * SEQ * D_MODEL;

  // workspace carve-up (~88 MB)
  char* p = (char*)d_ws;
  _Float16* xln = (_Float16*)p; p += (size_t)BATCH * SEQ * D_MODEL * 2;
  _Float16* wqT = (_Float16*)p; p += (size_t)D_MODEL * D_MODEL * 2;
  _Float16* wkT = (_Float16*)p; p += (size_t)D_MODEL * D_MODEL * 2;
  _Float16* wvT = (_Float16*)p; p += (size_t)D_MODEL * D_MODEL * 2;
  _Float16* woT = (_Float16*)p; p += (size_t)D_MODEL * D_MODEL * 2;
  _Float16* Qb  = (_Float16*)p; p += (size_t)BATCH * SEQ * D_MODEL * 2;
  _Float16* Kb  = (_Float16*)p; p += (size_t)BATCH * SEQ * D_MODEL * 2;
  _Float16* Vb  = (_Float16*)p; p += (size_t)BATCH * SEQ * D_MODEL * 2;
  _Float16* Ctx = (_Float16*)p; p += (size_t)BATCH * SEQ * D_MODEL * 2;
  float* table  = (float*)p;    p += (size_t)NHEADS * 4096 * 4;

  ln_kernel<<<BATCH * SEQ, 256, 0, stream>>>(hs, lnw, xln);
  wt_kernel<<<1024, 256, 0, stream>>>(Wq, wqT);
  wt_kernel<<<1024, 256, 0, stream>>>(Wk, wkT);
  wt_kernel<<<1024, 256, 0, stream>>>(Wv, wvT);
  wt_kernel<<<1024, 256, 0, stream>>>(Wo, woT);
  bias_table_kernel<<<(NHEADS * 4096) / 256, 256, 0, stream>>>(rel, table);
  bias_out_kernel<<<65536, 256, 0, stream>>>(table, bias_out);

  dim3 gg(64, 8);  // M/128 x N/128
  gemm_f16_kernel<<<gg, 256, 0, stream>>>(xln, wqT, Qb, BATCH * SEQ, D_MODEL, D_MODEL);
  gemm_f16_kernel<<<gg, 256, 0, stream>>>(xln, wkT, Kb, BATCH * SEQ, D_MODEL, D_MODEL);
  gemm_f16_kernel<<<gg, 256, 0, stream>>>(xln, wvT, Vb, BATCH * SEQ, D_MODEL, D_MODEL);

  attn_kernel<<<dim3(SEQ / 128, NHEADS, BATCH), 128, 0, stream>>>(Qb, Kb, Vb, table, Ctx);

  gemm_out_kernel<<<gg, 256, 0, stream>>>(Ctx, woT, hs, out, BATCH * SEQ, D_MODEL, D_MODEL);
}
